// self_attention_memory_module_75823352644193
// MI455X (gfx1250) — matrix operand order repaired and checked by reading, not compiled, not run on hardware
//
#include <hip/hip_runtime.h>
#include <hip/hip_bf16.h>

// ---------------------------------------------------------------------------
// SA-ConvLSTM cell on MI455X (gfx1250).
// All channel-mixing as bf16 WMMA GEMMs with spatial-major ("transposed")
// activation layout so every WMMA operand is two contiguous b128 loads/lane.
// GEMM wave tile 32x32 -> 2 b128 loads per v_wmma_f32_16x16x32_bf16.
// Attention: flash-style online softmax, 16-way A-reuse in the PV phase.
// ---------------------------------------------------------------------------

typedef __attribute__((ext_vector_type(16))) __bf16 v16bf;
typedef __attribute__((ext_vector_type(8)))  float  v8f;
typedef unsigned short u16;
typedef unsigned int   u32;

#define HWD   1024      // H*W
#define BATCH 32
#define CDIM  256
#define HID   128

__device__ __forceinline__ u16 f2bf(float f) {
    u32 u = __builtin_bit_cast(u32, f);
    u32 r = (u + 0x7FFFu + ((u >> 16) & 1u)) >> 16;   // round-to-nearest-even
    return (u16)r;
}

struct B32B { uint4 lo, hi; };

// 16 contiguous bf16 at p (16B-aligned) -> v16bf  (two b128 loads)
__device__ __forceinline__ v16bf ld_contig16(const u16* p) {
    B32B t;
    t.lo = *(const uint4*)(p);
    t.hi = *(const uint4*)(p + 8);
    return __builtin_bit_cast(v16bf, t);
}
// A-tile fragment: row pointer (K-contiguous); block [kh*16, +16)
__device__ __forceinline__ v16bf ld_a_frag(const u16* rowp, int kh) {
    B32B t;
    t.lo = *(const uint4*)(rowp + kh * 16);
    t.hi = *(const uint4*)(rowp + kh * 16 + 8);
    return __builtin_bit_cast(v16bf, t);
}
__device__ __forceinline__ uint4 pack_bf16x8(const float* e) {
    uint4 r;
    r.x = (u32)f2bf(e[0]) | ((u32)f2bf(e[1]) << 16);
    r.y = (u32)f2bf(e[2]) | ((u32)f2bf(e[3]) << 16);
    r.z = (u32)f2bf(e[4]) | ((u32)f2bf(e[5]) << 16);
    r.w = (u32)f2bf(e[6]) | ((u32)f2bf(e[7]) << 16);
    return r;
}

// ------------------------------- converters --------------------------------

__global__ void cvt_f32_bf16_kernel(const float* __restrict__ src,
                                    u16* __restrict__ dst, int n) {
    for (int i = blockIdx.x * blockDim.x + threadIdx.x; i < n;
         i += gridDim.x * blockDim.x)
        dst[i] = f2bf(src[i]);
}

// Transpose-convert activation [B, 256, HWD] f32 -> [B, HWD, 256] bf16,
// optionally duplicating into the [Z,h] concat buffer (stride 768, ch 512+).
__global__ __launch_bounds__(256)
void cvt_transpose_kernel(const float* __restrict__ src,
                          u16* __restrict__ dstT,      // [B, HWD, 256]
                          u16* __restrict__ combinT)   // [B, HWD, 768] or null
{
    __shared__ u16 tile[32][33];
    const int b  = blockIdx.z;
    const int p0 = blockIdx.x * 32;
    const int c0 = blockIdx.y * 32;
    const int tx = threadIdx.x & 31;
    const int ty = threadIdx.x >> 5;           // 0..7

    #pragma unroll
    for (int j = 0; j < 4; ++j) {
        int c = c0 + ty + 8 * j;
        tile[ty + 8 * j][tx] =
            f2bf(src[(size_t)b * CDIM * HWD + (size_t)c * HWD + p0 + tx]);
    }
    __syncthreads();
    #pragma unroll
    for (int j = 0; j < 4; ++j) {
        int p = p0 + ty + 8 * j;
        u16 v = tile[tx][ty + 8 * j];
        dstT[(size_t)b * HWD * CDIM + (size_t)p * CDIM + c0 + tx] = v;
        if (combinT)
            combinT[(size_t)b * HWD * 768 + (size_t)p * 768 + 512 + c0 + tx] = v;
    }
}

// ------------------------------ WMMA GEMM ----------------------------------
// Y[o,p] = sum_c W[o,c] * XT[p,c] + bias[o] for each batch image.
//   XT : [B, HWD, Cin] bf16 (spatial-major)
// Output modes:
//   Yf    != null : transposed f32   YT[b, p, ycofs+o]   (ystride channels)
//   chanM != 0    : channel-major bf16 Y[b, o, p]        (attention V)
//   else          : transposed bf16  YT[b, p, ycofs+o]   (ystride channels)
// Block: 256 threads = 8 waves; block tile 128(M) x 64(N); wave tile 32 x 32:
// per K-step 8 b128 loads feed 4 WMMAs (2 b128 / WMMA).
__global__ __launch_bounds__(256, 1)
void gemm_bf16_wmma_kernel(const u16* __restrict__ Wb,    // [Cout, Cin] bf16
                           const u16* __restrict__ XT,    // [B, HWD, Cin] bf16
                           const float* __restrict__ bias,// [Cout] f32
                           u16* __restrict__ Ybf,
                           float* __restrict__ Yf,
                           int Cout, int Cin,
                           int ystride, int ycofs, int chanM)
{
    const int b    = blockIdx.z;
    const int n0   = blockIdx.x * 64;
    const int m0b  = blockIdx.y * 128;
    const int tid  = threadIdx.x;
    const int wid  = tid >> 5;
    const int lane = tid & 31;
    const int wm   = wid >> 1;              // 0..3  (M step 32)
    const int wn   = wid & 1;               // 0..1  (N step 32)
    const int m0   = m0b + wm * 32;
    const int nn0  = n0 + wn * 32;
    const int ln   = lane & 15;
    const int kh   = lane >> 4;

    const u16* x0  = XT + (size_t)b * HWD * Cin + (size_t)(nn0 + ln) * Cin;
    const u16* x1  = x0 + (size_t)16 * Cin;
    const u16* wr0 = Wb + (size_t)(m0 + ln) * Cin;
    const u16* wr1 = wr0 + (size_t)16 * Cin;

    v8f acc00 = {}, acc01 = {}, acc10 = {}, acc11 = {};
    for (int k0 = 0; k0 < Cin; k0 += 32) {
        v16bf a0 = ld_a_frag(wr0 + k0, kh);           // rows m0   .. m0+15
        v16bf a1 = ld_a_frag(wr1 + k0, kh);           // rows m0+16.. m0+31
        v16bf b0 = ld_contig16(x0 + k0 + kh * 16);    // cols nn0  .. nn0+15
        v16bf b1 = ld_contig16(x1 + k0 + kh * 16);    // cols nn0+16..nn0+31
        acc00 = __builtin_amdgcn_wmma_f32_16x16x32_bf16(false, a0, false, b0,
                                                        (short)0, acc00, false, false);
        acc01 = __builtin_amdgcn_wmma_f32_16x16x32_bf16(false, a0, false, b1,
                                                        (short)0, acc01, false, false);
        acc10 = __builtin_amdgcn_wmma_f32_16x16x32_bf16(false, a1, false, b0,
                                                        (short)0, acc10, false, false);
        acc11 = __builtin_amdgcn_wmma_f32_16x16x32_bf16(false, a1, false, b1,
                                                        (short)0, acc11, false, false);
    }

    // D layout: lane=N, M = 8*kh + v  -> per lane 8 *consecutive* channels.
    float e00[8], e01[8], e10[8], e11[8];
    #pragma unroll
    for (int v = 0; v < 8; ++v) {
        float bv0 = bias ? bias[m0 + kh * 8 + v]      : 0.0f;
        float bv1 = bias ? bias[m0 + 16 + kh * 8 + v] : 0.0f;
        e00[v] = acc00[v] + bv0;  e10[v] = acc10[v] + bv1;
        e01[v] = acc01[v] + bv0;  e11[v] = acc11[v] + bv1;
    }

    if (Yf) {                       // transposed f32 out (float4 stores)
        size_t r0 = (size_t)b * HWD * ystride + (size_t)(nn0 + ln) * ystride
                  + ycofs + m0 + kh * 8;
        size_t r1 = r0 + (size_t)16 * ystride;
        *(float4*)(Yf + r0)      = make_float4(e00[0], e00[1], e00[2], e00[3]);
        *(float4*)(Yf + r0 + 4)  = make_float4(e00[4], e00[5], e00[6], e00[7]);
        *(float4*)(Yf + r0 + 16) = make_float4(e10[0], e10[1], e10[2], e10[3]);
        *(float4*)(Yf + r0 + 20) = make_float4(e10[4], e10[5], e10[6], e10[7]);
        *(float4*)(Yf + r1)      = make_float4(e01[0], e01[1], e01[2], e01[3]);
        *(float4*)(Yf + r1 + 4)  = make_float4(e01[4], e01[5], e01[6], e01[7]);
        *(float4*)(Yf + r1 + 16) = make_float4(e11[0], e11[1], e11[2], e11[3]);
        *(float4*)(Yf + r1 + 20) = make_float4(e11[4], e11[5], e11[6], e11[7]);
    } else if (chanM) {             // channel-major bf16 (attention V)
        #pragma unroll
        for (int v = 0; v < 8; ++v) {
            size_t o0 = (size_t)b * Cout * HWD
                      + (size_t)(m0 + kh * 8 + v) * HWD + nn0 + ln;
            size_t o1 = o0 + (size_t)16 * HWD;
            Ybf[o0]      = f2bf(e00[v]);
            Ybf[o0 + 16] = f2bf(e01[v]);
            Ybf[o1]      = f2bf(e10[v]);
            Ybf[o1 + 16] = f2bf(e11[v]);
        }
    } else {                        // transposed bf16 out (b128 stores)
        size_t r0 = (size_t)b * HWD * ystride + (size_t)(nn0 + ln) * ystride
                  + ycofs + m0 + kh * 8;
        size_t r1 = r0 + (size_t)16 * ystride;
        *(uint4*)(Ybf + r0)      = pack_bf16x8(e00);
        *(uint4*)(Ybf + r0 + 16) = pack_bf16x8(e10);
        *(uint4*)(Ybf + r1)      = pack_bf16x8(e01);
        *(uint4*)(Ybf + r1 + 16) = pack_bf16x8(e11);
    }
}

// --------------------------- flash attention -------------------------------
// Z[p, c] = sum_q softmax_q(Q^T K)[p,q] * V[c,q]
//   QT,KT : [B, HWD, 128] bf16 (spatial-major)  -> contiguous b128 fragments
//   V     : [B, 256, HWD] bf16 (channel-major)  -> contiguous along q
//   ZT    : [B, HWD, 512] bf16 (spatial-major), channels at zcofs
// 128 threads = 4 independent waves; wave owns a 16-row p-tile, streams q in
// chunks of 32, O (16x256) lives in 16 f32 accumulators, online softmax.
__global__ __launch_bounds__(128, 1)
void attn_wmma_kernel(const u16* __restrict__ QT,
                      const u16* __restrict__ KT,
                      const u16* __restrict__ V,
                      u16* __restrict__ ZT,
                      int zcofs)
{
    const int b    = blockIdx.y;
    const int tid  = threadIdx.x;
    const int wid  = tid >> 5;
    const int lane = tid & 31;
    const int ln   = lane & 15;
    const int kh   = lane >> 4;
    const int p0   = blockIdx.x * 64 + wid * 16;

    const u16* Qp = QT + (size_t)b * HWD * HID + (size_t)(p0 + ln) * HID;
    const u16* Kb = KT + (size_t)b * HWD * HID;
    const u16* Vb = V + (size_t)b * CDIM * HWD;

    __shared__ alignas(16) u16 plds[4][16 * 32];   // per-wave P scratch (bf16)
    u16* pl = plds[wid];

    // Preload Q^T A-fragments: 4 chunks of 32 along d (b128 loads).
    v16bf qa[4];
    #pragma unroll
    for (int kc = 0; kc < 4; ++kc) qa[kc] = ld_a_frag(Qp + kc * 32, kh);

    float mrow[8], lrow[8];
    #pragma unroll
    for (int v = 0; v < 8; ++v) { mrow[v] = -3.0e38f; lrow[v] = 0.0f; }
    v8f o[16];
    #pragma unroll
    for (int t = 0; t < 16; ++t) o[t] = (v8f){};

    for (int q0 = 0; q0 < HWD; q0 += 32) {
        // ---- S = Q^T K : 16x32 tile, K-fragments contiguous from KT ----
        const u16* k0p = Kb + (size_t)(q0 + ln) * HID + kh * 16;
        const u16* k1p = k0p + (size_t)16 * HID;
        v8f s0 = {}, s1 = {};
        #pragma unroll
        for (int kc = 0; kc < 4; ++kc) {
            v16bf b0 = ld_contig16(k0p + kc * 32);
            v16bf b1 = ld_contig16(k1p + kc * 32);
            s0 = __builtin_amdgcn_wmma_f32_16x16x32_bf16(false, qa[kc], false, b0,
                                                         (short)0, s0, false, false);
            s1 = __builtin_amdgcn_wmma_f32_16x16x32_bf16(false, qa[kc], false, b1,
                                                         (short)0, s1, false, false);
        }

        // ---- online softmax over this q-chunk (rows M = 8*kh + v) ----
        float alpha[8];
        #pragma unroll
        for (int v = 0; v < 8; ++v) {
            float t = fmaxf(s0[v], s1[v]);
            t = fmaxf(t, __shfl_xor(t, 1, 32));
            t = fmaxf(t, __shfl_xor(t, 2, 32));
            t = fmaxf(t, __shfl_xor(t, 4, 32));
            t = fmaxf(t, __shfl_xor(t, 8, 32));
            float mnew = fmaxf(mrow[v], t);
            alpha[v] = __expf(mrow[v] - mnew);
            s0[v] = __expf(s0[v] - mnew);
            s1[v] = __expf(s1[v] - mnew);
            float rs = s0[v] + s1[v];
            rs += __shfl_xor(rs, 1, 32);
            rs += __shfl_xor(rs, 2, 32);
            rs += __shfl_xor(rs, 4, 32);
            rs += __shfl_xor(rs, 8, 32);
            lrow[v] = lrow[v] * alpha[v] + rs;
            mrow[v] = mnew;
        }
        #pragma unroll
        for (int t = 0; t < 16; ++t)
            #pragma unroll
            for (int v = 0; v < 8; ++v) o[t][v] *= alpha[v];

        // ---- stage P (bf16) via LDS: D-layout -> A-layout transpose ----
        #pragma unroll
        for (int v = 0; v < 8; ++v) {
            int M = kh * 8 + v;
            pl[M * 32 + ln]      = f2bf(s0[v]);
            pl[M * 32 + 16 + ln] = f2bf(s1[v]);
        }
        asm volatile("s_wait_dscnt 0" ::: "memory");   // wave-private RAW fence
        v16bf pa = ld_a_frag(pl + ln * 32, kh);        // two ds_load_b128

        // ---- O += P * V^T : 16 c-tiles, V rows contiguous along q ----
        #pragma unroll
        for (int t = 0; t < 16; ++t) {
            v16bf vb = ld_contig16(Vb + (size_t)(t * 16 + ln) * HWD + q0 + kh * 16);
            o[t] = __builtin_amdgcn_wmma_f32_16x16x32_bf16(false, pa, false, vb,
                                                           (short)0, o[t], false, false);
        }
    }

    // ---- epilogue: normalize, scatter ZT[b, p, zcofs+c] ----
    #pragma unroll
    for (int t = 0; t < 16; ++t) {
        #pragma unroll
        for (int v = 0; v < 8; ++v) {
            int p = p0 + kh * 8 + v;
            int c = t * 16 + ln;
            ZT[(size_t)b * HWD * 512 + (size_t)p * 512 + zcofs + c] =
                f2bf(o[t][v] / lrow[v]);
        }
    }
}

// ------------------------------- gating ------------------------------------
// combT: [B, HWD, 768] f32 (spatial-major).  Indexed channel-fastest so the
// comb reads are fully coalesced; m / out use the reference [B,C,H,W] layout.
__global__ void gate_kernel(const float* __restrict__ combT,
                            const float* __restrict__ m,
                            float* __restrict__ out, int n)
{
    for (int i = blockIdx.x * blockDim.x + threadIdx.x; i < n;
         i += gridDim.x * blockDim.x) {
        int b   = i >> 18;                 // 1024*256 per batch
        int rem = i & 262143;
        int p   = rem >> 8;
        int c   = rem & 255;
        size_t base = (size_t)b * HWD * 768 + (size_t)p * 768;
        float mo = combT[base + c];
        float mg = combT[base + 256 + c];
        float mi = combT[base + 512 + c];
        float smi = 1.0f / (1.0f + __expf(-mi));
        size_t oidx = (size_t)b * (CDIM * HWD) + (size_t)c * HWD + p;
        float new_m = (1.0f - smi) * m[oidx] + smi * tanhf(mg);
        float new_h = (1.0f / (1.0f + __expf(-mo))) * new_m;
        out[oidx]     = new_h;
        out[n + oidx] = new_m;
    }
}

// ------------------------------- launcher ----------------------------------
extern "C" void kernel_launch(void* const* d_in, const int* in_sizes, int n_in,
                              void* d_out, int out_size, void* d_ws, size_t ws_size,
                              hipStream_t stream)
{
    const float* h   = (const float*)d_in[0];
    const float* m   = (const float*)d_in[1];
    const float* Wq  = (const float*)d_in[2];  const float* bq  = (const float*)d_in[3];
    const float* Wk  = (const float*)d_in[4];  const float* bk  = (const float*)d_in[5];
    const float* Wv  = (const float*)d_in[6];  const float* bv  = (const float*)d_in[7];
    const float* Wk2 = (const float*)d_in[8];  const float* bk2 = (const float*)d_in[9];
    const float* Wv2 = (const float*)d_in[10]; const float* bv2 = (const float*)d_in[11];
    const float* Wz  = (const float*)d_in[12]; const float* bz  = (const float*)d_in[13];
    const float* Wm  = (const float*)d_in[14]; const float* bm  = (const float*)d_in[15];

    char* ws = (char*)d_ws;
    size_t off = 0;
    auto alloc = [&](size_t bytes) -> char* {
        char* p = ws + off;
        off = (off + bytes + 255) & ~(size_t)255;
        return p;
    };

    const size_t NACT = (size_t)BATCH * CDIM * HWD;        // 8,388,608

    u16* wq_bf  = (u16*)alloc(128 * 256 * 2);
    u16* wk_bf  = (u16*)alloc(128 * 256 * 2);
    u16* wk2_bf = (u16*)alloc(128 * 256 * 2);
    u16* wv_bf  = (u16*)alloc(256 * 256 * 2);
    u16* wv2_bf = (u16*)alloc(256 * 256 * 2);
    u16* wz_bf  = (u16*)alloc(512 * 512 * 2);
    u16* wm_bf  = (u16*)alloc(768 * 768 * 2);
    u16* h_bfT  = (u16*)alloc(NACT * 2);                        // [B,HWD,256]
    u16* m_bfT  = (u16*)alloc(NACT * 2);                        // [B,HWD,256]
    u16* combinT= (u16*)alloc((size_t)BATCH * HWD * 768 * 2);   // [B,HWD,768]

    // Pool: attention intermediates, later aliased by the f32 comb buffer.
    char* pool = ws + off;
    u16* q_bfT = (u16*)(pool);                     // [B,HWD,128]
    u16* kh_bfT= (u16*)(pool + 8388608);           // [B,HWD,128]
    u16* km_bfT= (u16*)(pool + 16777216);          // [B,HWD,128]
    u16* vh_bf = (u16*)(pool + 25165824);          // [B,256,HWD] channel-major
    u16* vm_bf = (u16*)(pool + 41943040);          // [B,256,HWD] channel-major
    u16* zcatT = (u16*)(pool + 58720256);          // [B,HWD,512]
    float* combT_f32 = (float*)pool;               // [B,HWD,768], aliases pool

    dim3 blk256(256), blk128(128);

    // ---- phase 1: f32 -> bf16 conversions (+ activation transpose) ----
    cvt_f32_bf16_kernel<<<128, blk256, 0, stream>>>(Wq,  wq_bf,  128 * 256);
    cvt_f32_bf16_kernel<<<128, blk256, 0, stream>>>(Wk,  wk_bf,  128 * 256);
    cvt_f32_bf16_kernel<<<128, blk256, 0, stream>>>(Wk2, wk2_bf, 128 * 256);
    cvt_f32_bf16_kernel<<<256, blk256, 0, stream>>>(Wv,  wv_bf,  256 * 256);
    cvt_f32_bf16_kernel<<<256, blk256, 0, stream>>>(Wv2, wv2_bf, 256 * 256);
    cvt_f32_bf16_kernel<<<512, blk256, 0, stream>>>(Wz,  wz_bf,  512 * 512);
    cvt_f32_bf16_kernel<<<1024, blk256, 0, stream>>>(Wm, wm_bf,  768 * 768);
    cvt_transpose_kernel<<<dim3(32, 8, BATCH), blk256, 0, stream>>>(h, h_bfT, combinT);
    cvt_transpose_kernel<<<dim3(32, 8, BATCH), blk256, 0, stream>>>(m, m_bfT, nullptr);

    // ---- phase 2: projection GEMMs ----
    gemm_bf16_wmma_kernel<<<dim3(16, 1, BATCH), blk256, 0, stream>>>(
        wq_bf, h_bfT, bq, q_bfT, nullptr, 128, 256, 128, 0, 0);
    gemm_bf16_wmma_kernel<<<dim3(16, 1, BATCH), blk256, 0, stream>>>(
        wk_bf, h_bfT, bk, kh_bfT, nullptr, 128, 256, 128, 0, 0);
    gemm_bf16_wmma_kernel<<<dim3(16, 1, BATCH), blk256, 0, stream>>>(
        wk2_bf, m_bfT, bk2, km_bfT, nullptr, 128, 256, 128, 0, 0);
    gemm_bf16_wmma_kernel<<<dim3(16, 2, BATCH), blk256, 0, stream>>>(
        wv_bf, h_bfT, bv, vh_bf, nullptr, 256, 256, 256, 0, 1);   // channel-major
    gemm_bf16_wmma_kernel<<<dim3(16, 2, BATCH), blk256, 0, stream>>>(
        wv2_bf, m_bfT, bv2, vm_bf, nullptr, 256, 256, 256, 0, 1); // channel-major

    // ---- phase 3: attention branches (Z_h -> ch 0..255, Z_m -> 256..511) ----
    attn_wmma_kernel<<<dim3(16, BATCH), blk128, 0, stream>>>(
        q_bfT, kh_bfT, vh_bf, zcatT, 0);
    attn_wmma_kernel<<<dim3(16, BATCH), blk128, 0, stream>>>(
        q_bfT, km_bfT, vm_bf, zcatT, 256);

    // ---- phase 4: fuse GEMM Z = Wz*[Z_h;Z_m] + bz -> combinT ch 0..511 ----
    gemm_bf16_wmma_kernel<<<dim3(16, 4, BATCH), blk256, 0, stream>>>(
        wz_bf, zcatT, bz, combinT, nullptr, 512, 512, 768, 0, 0);

    // ---- phase 5: comb = Wm*[Z;h] + bm (transposed f32; aliases dead pool) ----
    gemm_bf16_wmma_kernel<<<dim3(16, 6, BATCH), blk256, 0, stream>>>(
        wm_bf, combinT, bm, nullptr, combT_f32, 768, 768, 768, 0, 0);

    // ---- phase 6: LSTM-style gating -> d_out = new_h ++ new_m ----
    gate_kernel<<<4096, blk256, 0, stream>>>(combT_f32, m, (float*)d_out, (int)NACT);

    (void)in_sizes; (void)n_in; (void)out_size; (void)ws_size;
}
